// BiometricLSTM_19189913878707
// MI455X (gfx1250) — compile-verified, matrix operand order checked by reading, not run on hardware
//
#include <hip/hip_runtime.h>

typedef __attribute__((ext_vector_type(16))) _Float16 v16h;
typedef __attribute__((ext_vector_type(8)))  float    v8f;

#define B_TOTAL 512
#define T_STEPS 2048
#define IN0     3
#define H       64
#define G4      256   // 4*H gate columns

// gfx1250 has a hardware v_tanh_f32 transcendental.
__device__ __forceinline__ float tanh_hw(float x) {
#if __has_builtin(__builtin_amdgcn_tanhf)
    return __builtin_amdgcn_tanhf(x);
#else
    // fallback: exp + fast rcp (v_rcp_f32), no IEEE divide chain
    float e = __expf(-2.0f * fabsf(x));
    float r = (1.0f - e) * __builtin_amdgcn_rcpf(1.0f + e);
    return copysignf(r, x);
#endif
}
// sigmoid via tanh: sigma(x) = 0.5*tanh(x/2) + 0.5  (mul + tanh + fma)
__device__ __forceinline__ float sigmoid_hw(float x) {
    return fmaf(0.5f, tanh_hw(0.5f * x), 0.5f);
}

// Fused 2-layer LSTM, persistent over T. One block = 16 batch rows, 16 waves.
// Wave w owns gate columns [16w, 16w+16); WMMA B-fragments of Whh0/Wih1/Whh1
// are preloaded into registers once and reused for all 2048 timesteps.
__global__ __launch_bounds__(512)
void lstm2_fused_kernel(const float* __restrict__ x,
                        const float* __restrict__ Wih0, const float* __restrict__ Whh0,
                        const float* __restrict__ bih0, const float* __restrict__ bhh0,
                        const float* __restrict__ Wih1, const float* __restrict__ Whh1,
                        const float* __restrict__ bih1, const float* __restrict__ bhh1,
                        float* __restrict__ out)
{
    __shared__ __align__(16) float     gbuf[16 * G4];  // 16 KB gate staging (f32)
    __shared__ __align__(16) _Float16  hA[16 * H];     // layer0 hidden (f16, row-major)
    __shared__ __align__(16) _Float16  hB[16 * H];     // layer1 hidden (f16, row-major)
    __shared__               float     xs[16 * IN0];   // x_t staging

    const int tid   = threadIdx.x;
    const int wave  = tid >> 5;        // 0..15
    const int lane  = tid & 31;
    const int bbase = blockIdx.x * 16; // global batch row base

    // Fragment geometry (16-bit A/B layout, 16x16x32):
    //  lanes 0..15 : row = lane,    K halves {0..7, 16..23} (+32*kb)
    //  lanes 16..31: row = lane-16, K halves {8..15, 24..31} (+32*kb)
    const int nrow = lane & 15;
    const int koff = (lane >> 4) ? 8 : 0;

    // ---- Preload weight B-fragments (gate col = 16*wave + nrow, K-contiguous
    //      rows of W directly give N-major/K-contiguous B^T). Held in VGPRs.
    v16h wB[6]; // [0,1]=Whh0 k-halves, [2,3]=Wih1, [4,5]=Whh1
    {
        const float* Wm0 = Whh0 + (wave * 16 + nrow) * H;
        const float* Wm1 = Wih1 + (wave * 16 + nrow) * H;
        const float* Wm2 = Whh1 + (wave * 16 + nrow) * H;
        #pragma unroll
        for (int kb = 0; kb < 2; ++kb) {
            const int k0 = kb * 32 + koff;
            v16h f0, f1, f2;
            #pragma unroll
            for (int i = 0; i < 8; ++i) {
                f0[i]     = (_Float16)Wm0[k0 + i];
                f0[8 + i] = (_Float16)Wm0[k0 + 16 + i];
                f1[i]     = (_Float16)Wm1[k0 + i];
                f1[8 + i] = (_Float16)Wm1[k0 + 16 + i];
                f2[i]     = (_Float16)Wm2[k0 + i];
                f2[8 + i] = (_Float16)Wm2[k0 + 16 + i];
            }
            wB[0 + kb] = f0;
            wB[2 + kb] = f1;
            wB[4 + kb] = f2;
        }
    }

    // ---- Pointwise thread ownership: (b = pb, j = pj) and (b = pb+8, j = pj)
    const int pb = tid >> 6;   // 0..7
    const int pj = tid & 63;   // hidden index
    float bs0[4], bs1[4], wx0[4][3];
    #pragma unroll
    for (int gt = 0; gt < 4; ++gt) {             // PyTorch gate order i,f,g,o
        const int gi = gt * H + pj;
        bs0[gt] = bih0[gi] + bhh0[gi];
        bs1[gt] = bih1[gi] + bhh1[gi];
        #pragma unroll
        for (int k = 0; k < IN0; ++k) wx0[gt][k] = Wih0[gi * IN0 + k];
    }

    float cA[2] = {0.f, 0.f};   // layer0 cell state for rows pb, pb+8
    float cB[2] = {0.f, 0.f};   // layer1 cell state
    float hFin[2] = {0.f, 0.f}; // final layer1 hidden

    for (int i = tid; i < 16 * H; i += 512) { hA[i] = (_Float16)0.f; hB[i] = (_Float16)0.f; }
    __syncthreads();

    // A-fragment load from a row-major 16xH f16 LDS buffer (aligned b128 pairs)
    auto loadA = [&](const _Float16* hbuf, int kb) -> v16h {
        v16h a;
        const _Float16* p = hbuf + nrow * H + kb * 32 + koff;
        #pragma unroll
        for (int i = 0; i < 8; ++i) { a[i] = p[i]; a[8 + i] = p[16 + i]; }
        return a;
    };
    // D-tile (16x16 f32) scatter to gate buffer: lane half selects M base
    const int mofs = (lane >> 4) ? 8 : 0;
    auto storeD = [&](v8f d) {
        float* gp = gbuf + mofs * G4 + wave * 16 + nrow;
        #pragma unroll
        for (int r = 0; r < 8; ++r) gp[r * G4] = d[r];
    };

    const v8f zero8 = {};

    for (int t = 0; t < T_STEPS; ++t) {
        // ---------- phase 1: gates0 = hA @ Whh0^T (K=64 -> 2 WMMA) ----------
        {
            v16h a0 = loadA(hA, 0);
            v16h a1 = loadA(hA, 1);
            v8f acc = __builtin_amdgcn_wmma_f32_16x16x32_f16(false, a0, false, wB[0],
                                                             (short)0, zero8, false, false);
            acc = __builtin_amdgcn_wmma_f32_16x16x32_f16(false, a1, false, wB[1],
                                                         (short)0, acc, false, false);
            storeD(acc);
        }
        if (tid < 16 * IN0) { // stage x_t (16 rows x 3)
            const int bb = tid / IN0, k = tid % IN0;
            xs[tid] = x[((size_t)(bbase + bb) * T_STEPS + t) * IN0 + k];
        }
        __syncthreads();

        // ---------- pointwise layer 0 (adds bias + K=3 input projection) ----
        #pragma unroll
        for (int hh = 0; hh < 2; ++hh) {
            const int b = pb + hh * 8;
            const float x0 = xs[b * IN0 + 0], x1 = xs[b * IN0 + 1], x2 = xs[b * IN0 + 2];
            const float* gr = gbuf + b * G4;
            float gi = gr[0 * H + pj] + bs0[0] + x0 * wx0[0][0] + x1 * wx0[0][1] + x2 * wx0[0][2];
            float gf = gr[1 * H + pj] + bs0[1] + x0 * wx0[1][0] + x1 * wx0[1][1] + x2 * wx0[1][2];
            float gg = gr[2 * H + pj] + bs0[2] + x0 * wx0[2][0] + x1 * wx0[2][1] + x2 * wx0[2][2];
            float go = gr[3 * H + pj] + bs0[3] + x0 * wx0[3][0] + x1 * wx0[3][1] + x2 * wx0[3][2];
            const float ig = sigmoid_hw(gi);
            const float fg = sigmoid_hw(gf);
            const float gc = tanh_hw(gg);
            const float og = sigmoid_hw(go);
            cA[hh] = fg * cA[hh] + ig * gc;
            const float h0 = og * tanh_hw(cA[hh]);
            hA[b * H + pj] = (_Float16)h0;
        }
        __syncthreads(); // gbuf reads done + hA(new) visible

        // ---------- phase 2: gates1 = hA_new @ Wih1^T + hB @ Whh1^T ----------
        {
            v16h a0 = loadA(hA, 0);
            v16h a1 = loadA(hA, 1);
            v16h b0 = loadA(hB, 0);
            v16h b1 = loadA(hB, 1);
            v8f acc = __builtin_amdgcn_wmma_f32_16x16x32_f16(false, a0, false, wB[2],
                                                             (short)0, zero8, false, false);
            acc = __builtin_amdgcn_wmma_f32_16x16x32_f16(false, a1, false, wB[3],
                                                         (short)0, acc, false, false);
            acc = __builtin_amdgcn_wmma_f32_16x16x32_f16(false, b0, false, wB[4],
                                                         (short)0, acc, false, false);
            acc = __builtin_amdgcn_wmma_f32_16x16x32_f16(false, b1, false, wB[5],
                                                         (short)0, acc, false, false);
            storeD(acc);
        }
        __syncthreads();

        // ---------- pointwise layer 1 ----------
        #pragma unroll
        for (int hh = 0; hh < 2; ++hh) {
            const int b = pb + hh * 8;
            const float* gr = gbuf + b * G4;
            float gi = gr[0 * H + pj] + bs1[0];
            float gf = gr[1 * H + pj] + bs1[1];
            float gg = gr[2 * H + pj] + bs1[2];
            float go = gr[3 * H + pj] + bs1[3];
            const float ig = sigmoid_hw(gi);
            const float fg = sigmoid_hw(gf);
            const float gc = tanh_hw(gg);
            const float og = sigmoid_hw(go);
            cB[hh] = fg * cB[hh] + ig * gc;
            const float h1 = og * tanh_hw(cB[hh]);
            hB[b * H + pj] = (_Float16)h1;
            hFin[hh] = h1;
        }
        __syncthreads(); // next iter reads hB / overwrites gbuf
    }

    out[(size_t)(bbase + pb) * H + pj]     = hFin[0];
    out[(size_t)(bbase + pb + 8) * H + pj] = hFin[1];
}

extern "C" void kernel_launch(void* const* d_in, const int* in_sizes, int n_in,
                              void* d_out, int out_size, void* d_ws, size_t ws_size,
                              hipStream_t stream) {
    (void)in_sizes; (void)n_in; (void)d_ws; (void)ws_size; (void)out_size;
    const float* x    = (const float*)d_in[0];
    const float* Wih0 = (const float*)d_in[1];
    const float* Whh0 = (const float*)d_in[2];
    const float* bih0 = (const float*)d_in[3];
    const float* bhh0 = (const float*)d_in[4];
    const float* Wih1 = (const float*)d_in[5];
    const float* Whh1 = (const float*)d_in[6];
    const float* bih1 = (const float*)d_in[7];
    const float* bhh1 = (const float*)d_in[8];
    float* out = (float*)d_out;

    dim3 grid(B_TOTAL / 16); // 32 batch tiles
    dim3 block(512);         // 16 wave32 waves
    lstm2_fused_kernel<<<grid, block, 0, stream>>>(x, Wih0, Whh0, bih0, bhh0,
                                                   Wih1, Whh1, bih1, bhh1, out);
}
